// PureTorchChamfer_67413806678291
// MI455X (gfx1250) — compile-verified, matrix operand order checked by reading, not run on hardware
//
#include <hip/hip_runtime.h>
#include <hip/hip_bf16.h>

typedef __attribute__((ext_vector_type(16))) _Float16 v16h;
typedef __attribute__((ext_vector_type(8)))  _Float16 v8h;
typedef __attribute__((ext_vector_type(4)))  _Float16 v4h;
typedef __attribute__((ext_vector_type(8)))  float    v8f;

#define BATCH 8
#define NPTS  4096
#define MPTS  4096
#define DIM   128
#define TI    128        // x rows per workgroup
#define TJ    256        // y rows per workgroup
#define LDS_STRIDE 136   // 128 + 8 halves padding -> 272B row stride, conflict-free b128 LDS loads

#define XS_HALFS  (TI * LDS_STRIDE)               // 17408 halves
#define YS_HALFS  (TJ * LDS_STRIDE)               // 34816 halves
#define F16_BYTES ((XS_HALFS + YS_HALFS) * 2)     // 104448 B
#define STAGE_BYTES ((TI + TJ) * DIM * 4)         // 196608 B fp32 staging
#define SHMEM_BYTES (F16_BYTES + STAGE_BYTES)     // 301056 B <= 320KB/WGP

union FragU { v16h v; v8h h[2]; };

#if defined(__HIP_DEVICE_COMPILE__) && __has_builtin(__builtin_amdgcn_global_load_async_to_lds_b128)
#define ASYNC_STAGE 1
#else
#define ASYNC_STAGE 0
#endif

#if ASYNC_STAGE
// Exact parameter types per clang diagnostic:
//   param0: int __vector(4) addrspace(1)*   (printed as "int __device__ *")
//   param1: int __vector(4) addrspace(3)*
typedef int async_data_t __attribute__((__vector_size__(4 * sizeof(int))));
typedef __attribute__((address_space(1))) async_data_t* glb_async_ptr;
typedef __attribute__((address_space(3))) async_data_t* lds_async_ptr;

// Flat pointers into the LDS aperture carry the LDS byte offset in addr[31:0]
// (ISA 10.2 aperture rules), so numeric truncation yields the AS(3) pointer.
__device__ __forceinline__ lds_async_ptr lds_cast(void* p) {
  return (lds_async_ptr)(unsigned int)(unsigned long long)(uintptr_t)p;
}
__device__ __forceinline__ glb_async_ptr glb_cast(const void* p) {
  return (glb_async_ptr)(unsigned long long)(uintptr_t)p;
}
__device__ __forceinline__ void wait_async0() {
#if __has_builtin(__builtin_amdgcn_s_wait_asynccnt)
  __builtin_amdgcn_s_wait_asynccnt(0);
#else
  asm volatile("s_wait_asynccnt 0" ::: "memory");
#endif
}
#endif

// ---------------------------------------------------------------------------
// 1) init row/col min buffers to +inf (bit pattern, distances are >= 0)
// ---------------------------------------------------------------------------
__global__ void init_min_kernel(unsigned* __restrict__ minbuf, int n) {
  int i = blockIdx.x * blockDim.x + threadIdx.x;
  if (i < n) minbuf[i] = 0x7F800000u;
}

// ---------------------------------------------------------------------------
// 2) fp32 squared row norms: one wave32 per row (128 floats = 4 per lane)
// ---------------------------------------------------------------------------
__global__ void row_norm_kernel(const float* __restrict__ x,
                                const float* __restrict__ y,
                                float* __restrict__ x2,
                                float* __restrict__ y2) {
  int row  = blockIdx.x;
  int lane = threadIdx.x;
  const float* src;
  float* dst;
  if (row < BATCH * NPTS) {
    src = x + (size_t)row * DIM;
    dst = x2 + row;
  } else {
    int r = row - BATCH * NPTS;
    src = y + (size_t)r * DIM;
    dst = y2 + r;
  }
  float4 v = ((const float4*)src)[lane];
  float s = v.x * v.x + v.y * v.y + v.z * v.z + v.w * v.w;
  s += __shfl_xor(s, 16, 32);
  s += __shfl_xor(s,  8, 32);
  s += __shfl_xor(s,  4, 32);
  s += __shfl_xor(s,  2, 32);
  s += __shfl_xor(s,  1, 32);
  if (lane == 0) *dst = s;
}

// ---------------------------------------------------------------------------
// 3) fused WMMA distance-tile + min-reduction kernel
//    grid: (M/256, N/128, B), block: 256 (8 waves as 2x4; each wave computes
//    a 64x64 output patch = 4x4 fragments of 16x16 -> 16 wmma : 16 ds_load)
// ---------------------------------------------------------------------------
__launch_bounds__(256)
__global__ void chamfer_tile_kernel(const float* __restrict__ x,
                                    const float* __restrict__ y,
                                    const float* __restrict__ x2,
                                    const float* __restrict__ y2,
                                    unsigned* __restrict__ rowmin,
                                    unsigned* __restrict__ colmin) {
  extern __shared__ char smem_raw[];
  _Float16* xs = (_Float16*)smem_raw;     // [TI][LDS_STRIDE] f16
  _Float16* ys = xs + XS_HALFS;           // [TJ][LDS_STRIDE] f16

  const int jt  = blockIdx.x;   // tile along M (y rows / output cols)
  const int it  = blockIdx.y;   // tile along N (x rows / output rows)
  const int b   = blockIdx.z;
  const int i0  = it * TI;
  const int j0  = jt * TJ;
  const int tid = threadIdx.x;

  const float* gx = x + ((size_t)b * NPTS + i0) * DIM;
  const float* gy = y + ((size_t)b * MPTS + j0) * DIM;

#if ASYNC_STAGE
  // ---- async global -> LDS fp32 staging (no VGPR round trip, ASYNCcnt) ----
  float* stX = (float*)(smem_raw + F16_BYTES);   // TI*DIM floats
  float* stY = stX + TI * DIM;                   // TJ*DIM floats
  for (int c = tid; c < TI * (DIM / 4); c += 256)
    __builtin_amdgcn_global_load_async_to_lds_b128(
        glb_cast(gx + (size_t)c * 4), lds_cast(stX + (size_t)c * 4), 0, 0);
  for (int c = tid; c < TJ * (DIM / 4); c += 256)
    __builtin_amdgcn_global_load_async_to_lds_b128(
        glb_cast(gy + (size_t)c * 4), lds_cast(stY + (size_t)c * 4), 0, 0);
  wait_async0();
  __syncthreads();
  // ---- LDS fp32 -> LDS f16 (padded WMMA tiles) ----
  for (int c = tid; c < TI * (DIM / 4); c += 256) {
    float4 v = *(const float4*)(stX + (size_t)c * 4);
    int r = c >> 5, kc = (c & 31) << 2;
    v4h hv = { (_Float16)v.x, (_Float16)v.y, (_Float16)v.z, (_Float16)v.w };
    *(v4h*)(xs + r * LDS_STRIDE + kc) = hv;
  }
  for (int c = tid; c < TJ * (DIM / 4); c += 256) {
    float4 v = *(const float4*)(stY + (size_t)c * 4);
    int r = c >> 5, kc = (c & 31) << 2;
    v4h hv = { (_Float16)v.x, (_Float16)v.y, (_Float16)v.z, (_Float16)v.w };
    *(v4h*)(ys + r * LDS_STRIDE + kc) = hv;
  }
#else
  // ---- fallback: global -> VGPR -> cvt -> LDS f16 ----
  for (int c = tid; c < TI * (DIM / 4); c += 256) {
    int r = c >> 5, kc = (c & 31) << 2;
    float4 v = *(const float4*)(gx + (size_t)r * DIM + kc);
    v4h hv = { (_Float16)v.x, (_Float16)v.y, (_Float16)v.z, (_Float16)v.w };
    *(v4h*)(xs + r * LDS_STRIDE + kc) = hv;
  }
  for (int c = tid; c < TJ * (DIM / 4); c += 256) {
    int r = c >> 5, kc = (c & 31) << 2;
    float4 v = *(const float4*)(gy + (size_t)r * DIM + kc);
    v4h hv = { (_Float16)v.x, (_Float16)v.y, (_Float16)v.z, (_Float16)v.w };
    *(v4h*)(ys + r * LDS_STRIDE + kc) = hv;
  }
#endif
  __syncthreads();

  const int wave = tid >> 5;
  const int lane = tid & 31;
  const int wr   = wave & 1;    // 2 wave-rows * 64 rows = 128
  const int wc   = wave >> 1;   // 4 wave-cols * 64 cols = 256
  const int h    = lane >> 4;   // half-wave selector (K striping)
  const int l16  = lane & 15;

  v8f zero = {};
  v8f acc[4][4];
#pragma unroll
  for (int mi = 0; mi < 4; ++mi)
#pragma unroll
    for (int nj = 0; nj < 4; ++nj) acc[mi][nj] = zero;

  // ---- K loop: 4 steps of 32 ----
#pragma unroll
  for (int ks = 0; ks < DIM; ks += 32) {
    FragU a[4], bb[4];
#pragma unroll
    for (int mi = 0; mi < 4; ++mi) {
      int row = wr * 64 + mi * 16 + l16;
      const _Float16* p = xs + row * LDS_STRIDE + ks + 8 * h;
      a[mi].h[0] = *(const v8h*)p;          // K = ks + 8h .. +7
      a[mi].h[1] = *(const v8h*)(p + 16);   // K = ks + 16 + 8h .. +7
    }
#pragma unroll
    for (int nj = 0; nj < 4; ++nj) {
      int row = wc * 64 + nj * 16 + l16;
      const _Float16* p = ys + row * LDS_STRIDE + ks + 8 * h;
      bb[nj].h[0] = *(const v8h*)p;
      bb[nj].h[1] = *(const v8h*)(p + 16);
    }
#pragma unroll
    for (int mi = 0; mi < 4; ++mi)
#pragma unroll
      for (int nj = 0; nj < 4; ++nj)
        acc[mi][nj] = __builtin_amdgcn_wmma_f32_16x16x32_f16(
            false, a[mi].v, false, bb[nj].v, (short)0, acc[mi][nj], false, false);
  }

  // ---- epilogue: dist = sqrt(max(x2 + y2 - 2*s, 0)); fused min reductions ----
  const float* x2b = x2 + (size_t)b * NPTS + i0;
  const float* y2b = y2 + (size_t)b * MPTS + j0;
  unsigned* rmg = rowmin + (size_t)b * NPTS + i0;
  unsigned* cmg = colmin + (size_t)b * MPTS + j0;

  float yj2[4];
#pragma unroll
  for (int nj = 0; nj < 4; ++nj) yj2[nj] = y2b[wc * 64 + nj * 16 + l16];

  float cmin[4] = { __builtin_inff(), __builtin_inff(),
                    __builtin_inff(), __builtin_inff() };

#pragma unroll
  for (int mi = 0; mi < 4; ++mi) {
#pragma unroll
    for (int vr = 0; vr < 8; ++vr) {
      int il = wr * 64 + mi * 16 + 8 * h + vr;   // C/D layout: M = vr + 8*(lane/16)
      float xi2  = x2b[il];
      float rmin = __builtin_inff();
#pragma unroll
      for (int nj = 0; nj < 4; ++nj) {
        float s  = acc[mi][nj][vr];
        float d2 = xi2 + yj2[nj] - 2.0f * s;
        float d  = __builtin_sqrtf(fmaxf(d2, 0.0f));
        rmin     = fminf(rmin, d);
        cmin[nj] = fminf(cmin[nj], d);
      }
      // reduce over the 16 lanes (columns) of this half-wave
      rmin = fminf(rmin, __shfl_xor(rmin, 1, 32));
      rmin = fminf(rmin, __shfl_xor(rmin, 2, 32));
      rmin = fminf(rmin, __shfl_xor(rmin, 4, 32));
      rmin = fminf(rmin, __shfl_xor(rmin, 8, 32));
      if (l16 == 0) atomicMin(rmg + il, __float_as_uint(rmin));
    }
  }

#pragma unroll
  for (int nj = 0; nj < 4; ++nj) {
    float cm = fminf(cmin[nj], __shfl_xor(cmin[nj], 16, 32)); // lanes L, L^16 share column
    if (h == 0) {
      int jl = wc * 64 + nj * 16 + l16;
      atomicMin(cmg + jl, __float_as_uint(cm));
    }
  }
}

// ---------------------------------------------------------------------------
// 4) final mean: (sum rowmin + sum colmin) / (B * 4096)
// ---------------------------------------------------------------------------
__global__ void final_reduce_kernel(const unsigned* __restrict__ minbuf,
                                    float* __restrict__ out) {
  __shared__ float red[256];
  float s = 0.0f;
  for (int i = threadIdx.x; i < 2 * BATCH * NPTS; i += 256)
    s += __uint_as_float(minbuf[i]);
  red[threadIdx.x] = s;
  __syncthreads();
  for (int w = 128; w > 0; w >>= 1) {
    if (threadIdx.x < w) red[threadIdx.x] += red[threadIdx.x + w];
    __syncthreads();
  }
  if (threadIdx.x == 0) out[0] = red[0] / (float)(BATCH * NPTS);
}

// ---------------------------------------------------------------------------
extern "C" void kernel_launch(void* const* d_in, const int* in_sizes, int n_in,
                              void* d_out, int out_size, void* d_ws, size_t ws_size,
                              hipStream_t stream) {
  (void)in_sizes; (void)n_in; (void)out_size; (void)ws_size;
  const float* x = (const float*)d_in[0];
  const float* y = (const float*)d_in[1];

  float* ws = (float*)d_ws;
  float*    x2     = ws;                         // 32768 floats
  float*    y2     = ws + 32768;                 // 32768 floats
  unsigned* rowmin = (unsigned*)(ws + 65536);    // 32768 uints
  unsigned* colmin = (unsigned*)(ws + 98304);    // 32768 uints (contiguous w/ rowmin)

  init_min_kernel<<<256, 256, 0, stream>>>(rowmin, 2 * BATCH * NPTS);
  row_norm_kernel<<<2 * BATCH * NPTS, 32, 0, stream>>>(x, y, x2, y2);

  dim3 grid(MPTS / TJ, NPTS / TI, BATCH);
  chamfer_tile_kernel<<<grid, 256, SHMEM_BYTES, stream>>>(x, y, x2, y2, rowmin, colmin);

  final_reduce_kernel<<<1, 256, 0, stream>>>(rowmin, (float*)d_out);
}
